// LongShortTermAttention_68771016344161
// MI455X (gfx1250) — compile-verified, matrix operand order checked by reading, not run on hardware
//
#include <hip/hip_runtime.h>
#include <hip/hip_bf16.h>

// ---------------------------------------------------------------------------
// LongShortTermAttention for MI455X (gfx1250).
// All GEMM-shaped work goes through v_wmma_f32_16x16x32_bf16 (wave32 WMMA).
// B=2, S=2048, D=1024, H=16, dh=64, NSEG=8.
// ---------------------------------------------------------------------------

typedef __attribute__((ext_vector_type(16))) __bf16 v16bf;
typedef __attribute__((ext_vector_type(8)))  __bf16 v8bf;
typedef __attribute__((ext_vector_type(8)))  float  v8f;

union AFrag { v16bf v; v8bf h[2]; };

#define WMMA_BF16(A, Bm, Cacc) \
    __builtin_amdgcn_wmma_f32_16x16x32_bf16(false, (A), false, (Bm), (short)0, (Cacc), false, false)

static constexpr int kB    = 2;
static constexpr int kS    = 2048;
static constexpr int kD    = 1024;
static constexpr int kH    = 16;
static constexpr int kDh   = 64;
static constexpr int kNSEG = 8;
static constexpr int kD3   = 3 * kD;       // 3072
static constexpr int kBS   = kB * kS;      // 4096
static constexpr int kBH   = kB * kH;      // 32

// ---------------------------------------------------------------------------
// Segmentation: seg[t] = argmax_p (x[t] . protos[p]); cnt[b][p] histogram.
// ---------------------------------------------------------------------------
__global__ void seg_kernel(const float* __restrict__ x,
                           const float* __restrict__ protos,
                           int* __restrict__ seg, int* __restrict__ cnt) {
    int tid  = threadIdx.x;
    int lane = tid & 31;
    int wave = tid >> 5;
    int tok  = blockIdx.x * 8 + wave;          // grid = BS/8
    const float* xr = x + (size_t)tok * kD;

    float acc[kNSEG];
#pragma unroll
    for (int p = 0; p < kNSEG; ++p) acc[p] = 0.0f;

    for (int d = lane; d < kD; d += 32) {
        float xv = xr[d];
#pragma unroll
        for (int p = 0; p < kNSEG; ++p) acc[p] += xv * protos[p * kD + d];
    }
#pragma unroll
    for (int p = 0; p < kNSEG; ++p)
        for (int m = 1; m < 32; m <<= 1) acc[p] += __shfl_xor(acc[p], m, 32);

    if (lane == 0) {
        int best = 0; float bv = acc[0];
#pragma unroll
        for (int p = 1; p < kNSEG; ++p)
            if (acc[p] > bv) { bv = acc[p]; best = p; }
        seg[tok] = best;
        int b = tok / kS;
        atomicAdd(&cnt[b * kNSEG + best], 1);
    }
}

// ---------------------------------------------------------------------------
// Tiled bf16 WMMA GEMM:  C[M,N] = A[M,K] * Bt[N,K]^T + bias
// Block tile 64x128, BK=32, 8 waves (2 x 4); each wave owns a 32x32 output
// (4 accumulators) so every staged fragment feeds 2 WMMAs (4 WMMAs per
// 8 ds_load_b128).  f32 inputs are converted to bf16 while staging into LDS.
// ---------------------------------------------------------------------------
static constexpr int BM = 64, BN = 128, BK = 32;

__device__ inline void store_out(float*  p, float v) { *p = v; }
__device__ inline void store_out(__bf16* p, float v) { *p = (__bf16)v; }

template <typename TA, typename TB, typename TOUT>
__global__ __launch_bounds__(256)
void gemm_wmma(const TA* __restrict__ A, const TB* __restrict__ Bt,
               const float* __restrict__ bias, TOUT* __restrict__ C,
               int M, int N, int K, int ldc) {
    __shared__ __bf16 sA[BM * BK];   // 4 KB
    __shared__ __bf16 sB[BN * BK];   // 8 KB

    int tid  = threadIdx.x;
    int lane = tid & 31;
    int wave = tid >> 5;
    int wm   = wave & 1;         // 2 row strips of 32
    int wn   = wave >> 1;        // 4 col strips of 32
    int row  = lane & 15;
    int grp  = lane >> 4;

    int blockM = blockIdx.y * BM;
    int blockN = blockIdx.x * BN;

    int r    = tid >> 2;         // 0..63
    int ccol = (tid & 3) * 8;    // 0,8,16,24

    v8f c00 = {}, c01 = {}, c10 = {}, c11 = {};

    for (int k0 = 0; k0 < K; k0 += BK) {
        // ---- stage A tile 64x32 (8 elems / thread) ----
        {
            const TA* src = A + (size_t)(blockM + r) * K + k0 + ccol;
            __bf16* dst = sA + r * BK + ccol;
#pragma unroll
            for (int i = 0; i < 8; ++i) dst[i] = (__bf16)(float)src[i];
            if (k0 + BK < K) __builtin_prefetch(src + BK, 0, 1);
        }
        // ---- stage B tile 128x32 (two 64-row halves) ----
#pragma unroll
        for (int hf = 0; hf < 2; ++hf) {
            const TB* src = Bt + (size_t)(blockN + hf * 64 + r) * K + k0 + ccol;
            __bf16* dst = sB + (hf * 64 + r) * BK + ccol;
#pragma unroll
            for (int i = 0; i < 8; ++i) dst[i] = (__bf16)(float)src[i];
            if (k0 + BK < K) __builtin_prefetch(src + BK, 0, 1);
        }
        __syncthreads();

        AFrag a0, a1, b0, b1;
        const v8bf* pa0 = (const v8bf*)(sA + (wm * 32 + row) * BK);
        const v8bf* pa1 = (const v8bf*)(sA + (wm * 32 + 16 + row) * BK);
        a0.h[0] = pa0[grp]; a0.h[1] = pa0[grp + 2];
        a1.h[0] = pa1[grp]; a1.h[1] = pa1[grp + 2];
        const v8bf* pb0 = (const v8bf*)(sB + (wn * 32 + row) * BK);
        const v8bf* pb1 = (const v8bf*)(sB + (wn * 32 + 16 + row) * BK);
        b0.h[0] = pb0[grp]; b0.h[1] = pb0[grp + 2];
        b1.h[0] = pb1[grp]; b1.h[1] = pb1[grp + 2];

        c00 = WMMA_BF16(a0.v, b0.v, c00);
        c01 = WMMA_BF16(a0.v, b1.v, c01);
        c10 = WMMA_BF16(a1.v, b0.v, c10);
        c11 = WMMA_BF16(a1.v, b1.v, c11);
        __syncthreads();
    }

    int m0 = blockM + wm * 32 + grp * 8;      // C layout: m = v + 8*grp
    int m1 = m0 + 16;
    int n0 = blockN + wn * 32 + row;          // n = lane%16
    int n1 = n0 + 16;
    float bv0 = bias ? bias[n0] : 0.0f;
    float bv1 = bias ? bias[n1] : 0.0f;
#pragma unroll
    for (int v = 0; v < 8; ++v) {
        store_out(C + (size_t)(m0 + v) * ldc + n0, c00[v] + bv0);
        store_out(C + (size_t)(m0 + v) * ldc + n1, c01[v] + bv1);
        store_out(C + (size_t)(m1 + v) * ldc + n0, c10[v] + bv0);
        store_out(C + (size_t)(m1 + v) * ldc + n1, c11[v] + bv1);
    }
}

// ---------------------------------------------------------------------------
// V transpose: Vt[bh][i][s] = qkv[(b*S+s)*3D + 2D + h*dh + i]  (bf16)
// ---------------------------------------------------------------------------
__global__ void vtrans_kernel(const __bf16* __restrict__ qkv,
                              __bf16* __restrict__ Vt) {
    size_t idx = (size_t)blockIdx.x * blockDim.x + threadIdx.x; // BH*dh*S total
    int s  = (int)(idx & (kS - 1));
    int i  = (int)((idx >> 11) & (kDh - 1));
    int bh = (int)(idx >> 17);
    int b  = bh >> 4, h = bh & 15;
    Vt[idx] = qkv[(size_t)(b * kS + s) * kD3 + 2 * kD + h * kDh + i];
}

// ---------------------------------------------------------------------------
// Flash attention (one 16-query tile per wave, 4 waves/block), 64-key chunks:
//   scores: 8x v_wmma per chunk (4 n-tiles x K split 32+32)
//   single online-softmax pass (shfl_xor 16-lane row reductions) per 64 keys
//   P -> wave-private LDS -> two 16x32 A fragments
//   P.V: 8x v_wmma per chunk against transposed V
// ---------------------------------------------------------------------------
template <bool MASKED>
__global__ __launch_bounds__(128)
void attn_kernel(const __bf16* __restrict__ qkv, const __bf16* __restrict__ Vt,
                 const int* __restrict__ seg, __bf16* __restrict__ ctx) {
    __shared__ __bf16 sP[4][16 * 64];   // 8 KB, wave-private 16x64 P tiles

    int tid  = threadIdx.x;
    int lane = tid & 31;
    int wave = tid >> 5;
    int row  = lane & 15;
    int grp  = lane >> 4;

    int bh = blockIdx.y;
    int b  = bh / kH, h = bh % kH;
    int qbase = blockIdx.x * 64 + wave * 16;

    // Q fragments (dh split into two 32-wide A fragments), loaded once
    const __bf16* qrow = qkv + (size_t)(b * kS + qbase + row) * kD3 + h * kDh;
    AFrag qa[2];
#pragma unroll
    for (int p = 0; p < 2; ++p) {
        const v8bf* pq = (const v8bf*)(qrow + p * 32);
        qa[p].h[0] = pq[grp];
        qa[p].h[1] = pq[grp + 2];
    }

    int segq[8];
    if (MASKED) {
#pragma unroll
        for (int v = 0; v < 8; ++v)
            segq[v] = seg[b * kS + qbase + grp * 8 + v];
    }

    float mstate[8], lstate[8];
#pragma unroll
    for (int v = 0; v < 8; ++v) { mstate[v] = -3.0e38f; lstate[v] = 0.0f; }
    v8f o[4] = {};

    __bf16* myP = sP[wave];
    const float scale = 0.125f;   // 1/sqrt(64)

    for (int kc = 0; kc < kS; kc += 64) {
        // ---- scores S = Q K^T for 64 keys (four 16x16 C tiles) ----
        v8f s[4] = {};
#pragma unroll
        for (int nt = 0; nt < 4; ++nt) {
            const __bf16* krow =
                qkv + (size_t)(b * kS + kc + nt * 16 + row) * kD3 + kD + h * kDh;
#pragma unroll
            for (int p = 0; p < 2; ++p) {
                AFrag kb;
                const v8bf* pk = (const v8bf*)(krow + p * 32);
                kb.h[0] = pk[grp];
                kb.h[1] = pk[grp + 2];
                s[nt] = WMMA_BF16(qa[p].v, kb.v, s[nt]);
            }
        }

        int segk[4];
        if (MASKED) {
#pragma unroll
            for (int nt = 0; nt < 4; ++nt)
                segk[nt] = seg[b * kS + kc + nt * 16 + row];
        }

        // ---- online softmax per row (row lives across a 16-lane group) ----
#pragma unroll
        for (int v = 0; v < 8; ++v) {
            float a[4];
#pragma unroll
            for (int nt = 0; nt < 4; ++nt) {
                a[nt] = s[nt][v] * scale;
                if (MASKED && segq[v] != segk[nt]) a[nt] = -1.0e9f;
            }
            float rm = fmaxf(fmaxf(a[0], a[1]), fmaxf(a[2], a[3]));
            for (int m2 = 1; m2 < 16; m2 <<= 1) rm = fmaxf(rm, __shfl_xor(rm, m2, 32));
            float mnew = fmaxf(mstate[v], rm);
            float corr = __expf(mstate[v] - mnew);
            float p[4], rs = 0.0f;
#pragma unroll
            for (int nt = 0; nt < 4; ++nt) { p[nt] = __expf(a[nt] - mnew); rs += p[nt]; }
            for (int m2 = 1; m2 < 16; m2 <<= 1) rs += __shfl_xor(rs, m2, 32);
            lstate[v] = lstate[v] * corr + rs;
            mstate[v] = mnew;
#pragma unroll
            for (int t = 0; t < 4; ++t) o[t][v] *= corr;

            int m = grp * 8 + v;                       // C-layout row
#pragma unroll
            for (int nt = 0; nt < 4; ++nt)
                myP[m * 64 + nt * 16 + row] = (__bf16)p[nt];
        }
        asm volatile("s_wait_dscnt 0x0" ::: "memory");  // in-wave LDS RAW fence

        // ---- reload P as two 16x32 A fragments ----
        AFrag pf[2];
        const v8bf* pp = (const v8bf*)(myP + row * 64);
#pragma unroll
        for (int pI = 0; pI < 2; ++pI) {
            pf[pI].h[0] = pp[pI * 4 + grp];
            pf[pI].h[1] = pp[pI * 4 + grp + 2];
        }

        // ---- O += P V  (V transposed: rows are dh, cols are keys) ----
#pragma unroll
        for (int t = 0; t < 4; ++t) {
            const __bf16* vrow = Vt + (size_t)(bh * kDh + t * 16 + row) * kS + kc;
            const v8bf* pv = (const v8bf*)vrow;
#pragma unroll
            for (int pI = 0; pI < 2; ++pI) {
                AFrag vb;
                vb.h[0] = pv[pI * 4 + grp];
                vb.h[1] = pv[pI * 4 + grp + 2];
                o[t] = WMMA_BF16(pf[pI].v, vb.v, o[t]);
            }
        }
    }

    // ---- normalize by row sum, store ctx[b*S+q][h*dh + c] ----
#pragma unroll
    for (int v = 0; v < 8; ++v) {
        int q = qbase + grp * 8 + v;
        float inv = 1.0f / lstate[v];
        size_t base = (size_t)(b * kS + q) * kD + h * kDh + row;
#pragma unroll
        for (int t = 0; t < 4; ++t)
            ctx[base + t * 16] = (__bf16)(o[t][v] * inv);
    }
}

// ---------------------------------------------------------------------------
// Singleton segments: short half of `combined` <- bf16(x)
// ---------------------------------------------------------------------------
__global__ void passthrough_kernel(const float* __restrict__ x,
                                   const int* __restrict__ seg,
                                   const int* __restrict__ cnt,
                                   __bf16* __restrict__ combined) {
    int tok = blockIdx.x;                  // BS blocks
    int b   = tok >> 11;
    if (cnt[b * kNSEG + seg[tok]] != 1) return;
    for (int d = threadIdx.x; d < kD; d += blockDim.x)
        combined[(size_t)tok * (2 * kD) + kD + d] = (__bf16)x[(size_t)tok * kD + d];
}

// ---------------------------------------------------------------------------
// Host-side launcher
// ---------------------------------------------------------------------------
static inline size_t align256(size_t v) { return (v + 255) & ~(size_t)255; }

extern "C" void kernel_launch(void* const* d_in, const int* in_sizes, int n_in,
                              void* d_out, int out_size, void* d_ws, size_t ws_size,
                              hipStream_t stream) {
    (void)in_sizes; (void)n_in; (void)out_size; (void)ws_size;

    const float* x        = (const float*)d_in[0];
    const float* lt_w_in  = (const float*)d_in[1];
    const float* lt_b_in  = (const float*)d_in[2];
    const float* lt_w_out = (const float*)d_in[3];
    const float* lt_b_out = (const float*)d_in[4];
    const float* st_w_in  = (const float*)d_in[5];
    const float* st_b_in  = (const float*)d_in[6];
    const float* st_w_out = (const float*)d_in[7];
    const float* st_b_out = (const float*)d_in[8];
    const float* protos   = (const float*)d_in[9];
    const float* fus_w    = (const float*)d_in[10];
    const float* fus_b    = (const float*)d_in[11];
    float* out = (float*)d_out;

    // ---- workspace carve-up (~59 MB) ----
    char* ws = (char*)d_ws;
    __bf16* qkv  = (__bf16*)ws; ws += align256((size_t)kBS * kD3 * 2);       // 24 MB
    __bf16* Vt   = (__bf16*)ws; ws += align256((size_t)kBH * kDh * kS * 2);  //  8 MB
    __bf16* ctx  = (__bf16*)ws; ws += align256((size_t)kBS * kD * 2);        //  8 MB
    __bf16* comb = (__bf16*)ws; ws += align256((size_t)kBS * 2 * kD * 2);    // 16 MB
    int*    seg  = (int*)ws;    ws += align256((size_t)kBS * 4);
    int*    cnt  = (int*)ws;    ws += align256((size_t)kB * kNSEG * 4);

    // segmentation (needed by masked attention + passthrough)
    hipMemsetAsync(cnt, 0, kB * kNSEG * sizeof(int), stream);
    seg_kernel<<<kBS / 8, 256, 0, stream>>>(x, protos, seg, cnt);

    dim3 blk256(256), blk128(128);
    dim3 g_qkv(kD3 / BN, kBS / BM);   // 24 x 64
    dim3 g_proj(kD / BN, kBS / BM);   //  8 x 64
    dim3 g_attn(kS / 64, kBH);        // 32 x 32

    // ---- long-term: full attention ----
    gemm_wmma<float, float, __bf16><<<g_qkv, blk256, 0, stream>>>(
        x, lt_w_in, lt_b_in, qkv, kBS, kD3, kD, kD3);
    vtrans_kernel<<<(kBH * kDh * kS) / 256, blk256, 0, stream>>>(qkv, Vt);
    attn_kernel<false><<<g_attn, blk128, 0, stream>>>(qkv, Vt, seg, ctx);
    gemm_wmma<__bf16, float, __bf16><<<g_proj, blk256, 0, stream>>>(
        ctx, lt_w_out, lt_b_out, comb, kBS, kD, kD, 2 * kD);          // left half

    // ---- short-term: segment-masked attention (reuses qkv/Vt/ctx) ----
    gemm_wmma<float, float, __bf16><<<g_qkv, blk256, 0, stream>>>(
        x, st_w_in, st_b_in, qkv, kBS, kD3, kD, kD3);
    vtrans_kernel<<<(kBH * kDh * kS) / 256, blk256, 0, stream>>>(qkv, Vt);
    attn_kernel<true><<<g_attn, blk128, 0, stream>>>(qkv, Vt, seg, ctx);
    gemm_wmma<__bf16, float, __bf16><<<g_proj, blk256, 0, stream>>>(
        ctx, st_w_out, st_b_out, comb + kD, kBS, kD, kD, 2 * kD);     // right half

    // singleton tokens pass raw x through on the short half
    passthrough_kernel<<<kBS, blk256, 0, stream>>>(x, seg, cnt, comb);

    // ---- fusion: out = combined @ fus_w.T + fus_b (f32 output) ----
    gemm_wmma<__bf16, float, float><<<g_proj, blk256, 0, stream>>>(
        comb, fus_w, fus_b, out, kBS, kD, 2 * kD, kD);
}
